// MultiheadAttention_44332652429499
// MI455X (gfx1250) — compile-verified, matrix operand order checked by reading, not run on hardware
//
#include <hip/hip_runtime.h>
#include <hip/hip_bf16.h>

// MI455X / gfx1250: wave32, WMMA 16x16x32 bf16 path.
// Workspace layout (requires >= 48MB):
//   Qw | Kw | Vw   each [B,H,S,64] bf16  (4*16*2048*64 * 2B = 16MB each)

typedef __attribute__((ext_vector_type(16))) __bf16 v16bf;
typedef __attribute__((ext_vector_type(8)))  float  v8f;

static constexpr int Bn = 4, Sn = 2048, Hn = 16, Dn = 64, Cn = 1024;

__device__ __forceinline__ __bf16 f2bf(float f) {
    unsigned u = __builtin_bit_cast(unsigned, f);
    unsigned r = (u + 0x7FFFu + ((u >> 16) & 1u)) >> 16;   // round-nearest-even
    unsigned short h = (unsigned short)r;
    return __builtin_bit_cast(__bf16, h);
}

__device__ __forceinline__ float rmax16(float v) {
    #pragma unroll
    for (int m = 1; m < 16; m <<= 1) v = fmaxf(v, __shfl_xor(v, m, 32));
    return v;
}
__device__ __forceinline__ float rsum16(float v) {
    #pragma unroll
    for (int m = 1; m < 16; m <<= 1) v += __shfl_xor(v, m, 32);
    return v;
}

// Async bulk copy: 16 bytes global -> LDS per lane, tracked by ASYNCcnt.
// GV addressing: VDST = LDS byte offset, VADDR = 64-bit global address.
__device__ __forceinline__ void async_g2lds_b128(unsigned lds_off, const void* gptr) {
    unsigned long long ga = (unsigned long long)(uintptr_t)gptr;
    asm volatile("global_load_async_to_lds_b128 %0, %1, off"
                 :: "v"(lds_off), "v"(ga) : "memory");
}
__device__ __forceinline__ void wait_asynccnt0() {
    asm volatile("s_wait_asynccnt 0x0" ::: "memory");
}

// ---------------------------------------------------------------------------
// Kernel 1: fused QKV projection  (x[8192,1024] @ W[1024,1024] + b) -> bf16
// blockIdx.z selects Q / K / V.  Output layout [B,H,S,64].
// ---------------------------------------------------------------------------
__global__ __launch_bounds__(256) void qkv_proj_kernel(
    const float* __restrict__ x,
    const float* __restrict__ Wq, const float* __restrict__ bq,
    const float* __restrict__ Wk, const float* __restrict__ bk,
    const float* __restrict__ Wv, const float* __restrict__ bv,
    __bf16* __restrict__ Qo, __bf16* __restrict__ Ko, __bf16* __restrict__ Vo)
{
    const float* W;  const float* bias;  __bf16* O;
    if      (blockIdx.z == 0) { W = Wq; bias = bq; O = Qo; }
    else if (blockIdx.z == 1) { W = Wk; bias = bk; O = Ko; }
    else                      { W = Wv; bias = bv; O = Vo; }

    __shared__ __bf16 As[128 * 36];   // A tile 128x32, padded stride 36
    __shared__ __bf16 Bs[64 * 36];    // B tile stored transposed [n][k]

    const int tid  = threadIdx.x;
    const int wid  = tid >> 5;
    const int lane = tid & 31;
    const int wm   = wid >> 1;        // 0..3 : 32-row slab
    const int wn   = wid & 1;         // 0..1 : 32-col slab
    const int m0   = blockIdx.x * 128;
    const int n0   = blockIdx.y * 64;
    const int lr   = lane & 15;
    const int koff = (lane >> 4) * 8;

    v8f acc[2][2] = {};

    for (int k0 = 0; k0 < Cn; k0 += 32) {
        __syncthreads();
        // A: 128x32 f32 -> bf16 LDS (coalesced), prefetch next k-chunk
        #pragma unroll
        for (int i = 0; i < 16; ++i) {
            int idx = tid + i * 256;
            int r = idx >> 5, c = idx & 31;
            const float* src = &x[(size_t)(m0 + r) * Cn + k0 + c];
            As[r * 36 + c] = f2bf(*src);
            if ((i & 3) == 0 && k0 + 32 < Cn) __builtin_prefetch(src + 32, 0, 1);
        }
        // B: 32x64 f32 -> transposed bf16 LDS Bs[n][k]
        #pragma unroll
        for (int i = 0; i < 8; ++i) {
            int idx = tid + i * 256;
            int kk = idx >> 6, nn = idx & 63;
            const float* src = &W[(size_t)(k0 + kk) * 1024 + n0 + nn];
            Bs[nn * 36 + kk] = f2bf(*src);
            if ((i & 3) == 0 && k0 + 32 < Cn) __builtin_prefetch(src + 32 * 1024, 0, 1);
        }
        __syncthreads();

        // Assemble fragments per the CDNA5 16-bit A/B VGPR layout.
        v16bf afr[2], bfr[2];
        #pragma unroll
        for (int r = 0; r < 2; ++r) {
            const int row = wm * 32 + r * 16 + lr;
            #pragma unroll
            for (int j = 0; j < 8; ++j) {
                afr[r][j]     = As[row * 36 + koff + j];
                afr[r][j + 8] = As[row * 36 + 16 + koff + j];
            }
        }
        #pragma unroll
        for (int c = 0; c < 2; ++c) {
            const int col = wn * 32 + c * 16 + lr;
            #pragma unroll
            for (int j = 0; j < 8; ++j) {
                bfr[c][j]     = Bs[col * 36 + koff + j];
                bfr[c][j + 8] = Bs[col * 36 + 16 + koff + j];
            }
        }
        #pragma unroll
        for (int r = 0; r < 2; ++r)
            #pragma unroll
            for (int c = 0; c < 2; ++c)
                acc[r][c] = __builtin_amdgcn_wmma_f32_16x16x32_bf16(
                    false, afr[r], false, bfr[c], (short)0, acc[r][c], false, false);
    }

    // Epilogue: + bias, bf16, scatter to [B,H,S,64]
    #pragma unroll
    for (int r = 0; r < 2; ++r) {
        #pragma unroll
        for (int c = 0; c < 2; ++c) {
            #pragma unroll
            for (int i = 0; i < 8; ++i) {
                const int row16 = (lane < 16) ? i : (i + 8);
                const int m = m0 + wm * 32 + r * 16 + row16;
                const int n = n0 + wn * 32 + c * 16 + lr;
                const float v = acc[r][c][i] + bias[n];
                const int bb = m >> 11;          // m / S
                const int ss = m & (Sn - 1);
                const int hh = n >> 6;           // n / 64
                const int dd = n & 63;
                O[(((size_t)(bb * Hn + hh) * Sn) + ss) * Dn + dd] = f2bf(v);
            }
        }
    }
}

// ---------------------------------------------------------------------------
// Kernel 2: flash attention.  One workgroup per (b*H+h, 128 q rows).
// Each wave owns 16 q rows.  KV chunks of 32, online softmax in f32.
// V tile staged via async global->LDS DMA (ASYNCcnt path).
// ---------------------------------------------------------------------------
__global__ __launch_bounds__(256) void flash_attn_kernel(
    const __bf16* __restrict__ Qw, const __bf16* __restrict__ Kw,
    const __bf16* __restrict__ Vw, float* __restrict__ out)
{
    __shared__ __bf16 Vs[32 * 66];        // V chunk [k][d], padded
    __shared__ __bf16 Ps[8 * 16 * 34];    // per-wave P tile [row][k], padded

    const int tid  = threadIdx.x;
    const int wid  = tid >> 5;
    const int lane = tid & 31;
    const int lr   = lane & 15;
    const int koff = (lane >> 4) * 8;

    const int bh = blockIdx.y;            // b*H + h
    const int b  = bh >> 4;
    const int h  = bh & 15;
    const size_t base = (size_t)bh * Sn * Dn;
    const int q0 = blockIdx.x * 128 + wid * 16;

    // Per-thread async copy slot: 8 bf16 (16 B).  kk = row in chunk, dd = col.
    const int vkk = tid >> 3;             // (tid*8)/64 : 0..31
    const int vdd = (tid & 7) * 8;        // (tid*8)%64
    const unsigned vs_off = (unsigned)(uintptr_t)(&Vs[vkk * 66 + vdd]);

    // Preload Q A-fragments (two 16x32 d-chunks)
    v16bf qa[2];
    #pragma unroll
    for (int c = 0; c < 2; ++c) {
        const size_t rbase = base + (size_t)(q0 + lr) * Dn + c * 32;
        #pragma unroll
        for (int j = 0; j < 8; ++j) {
            qa[c][j]     = Qw[rbase + koff + j];
            qa[c][j + 8] = Qw[rbase + 16 + koff + j];
        }
    }

    v8f acc[4] = {};
    float m_run[8], l_run[8];
    #pragma unroll
    for (int i = 0; i < 8; ++i) { m_run[i] = -1e30f; l_run[i] = 0.f; }

    const float scale = 0.125f;           // 1/sqrt(64)
    __bf16* myP = &Ps[wid * 16 * 34];

    for (int kv = 0; kv < Sn; kv += 32) {
        __syncthreads();                   // prior-iter Vs reads done
        // Async DMA: V chunk 32x64 bf16 -> LDS, one b128 per thread.
        async_g2lds_b128(vs_off, &Vw[base + (size_t)(kv + vkk) * Dn + vdd]);
        // Prefetch next chunk's K rows while DMA runs.
        if (kv + 32 < Sn)
            __builtin_prefetch(&Kw[base + (size_t)(kv + 32 + lr) * Dn], 0, 1);
        wait_asynccnt0();                  // own wave's DMA done
        __syncthreads();                   // all waves' DMA done

        // Scores: two 16x16 tiles (kv cols 0-15, 16-31); K^T B-fragments
        // read straight from global (row-major K is contiguous per lane).
        v8f sc[2] = {};
        #pragma unroll
        for (int t = 0; t < 2; ++t) {
            const size_t krow = base + (size_t)(kv + t * 16 + lr) * Dn;
            #pragma unroll
            for (int c = 0; c < 2; ++c) {
                v16bf kb;
                #pragma unroll
                for (int j = 0; j < 8; ++j) {
                    kb[j]     = Kw[krow + c * 32 + koff + j];
                    kb[j + 8] = Kw[krow + c * 32 + 16 + koff + j];
                }
                sc[t] = __builtin_amdgcn_wmma_f32_16x16x32_bf16(
                    false, qa[c], false, kb, (short)0, sc[t], false, false);
            }
        }

        // Online softmax (rows live per-(half,i); reduce across 16-lane half)
        float corr[8];
        #pragma unroll
        for (int i = 0; i < 8; ++i) {
            const float v0 = sc[0][i] * scale;
            const float v1 = sc[1][i] * scale;
            float t = rmax16(fmaxf(v0, v1));
            const float mn = fmaxf(m_run[i], t);
            const float p0 = __expf(v0 - mn);
            const float p1 = __expf(v1 - mn);
            const float ps = rsum16(p0 + p1);
            const float cr = __expf(m_run[i] - mn);
            l_run[i] = l_run[i] * cr + ps;
            m_run[i] = mn;
            corr[i]  = cr;
            const int row16 = (lane < 16) ? i : (i + 8);
            myP[row16 * 34 + lr]      = f2bf(p0);
            myP[row16 * 34 + 16 + lr] = f2bf(p1);
        }
        #pragma unroll
        for (int t = 0; t < 4; ++t)
            #pragma unroll
            for (int i = 0; i < 8; ++i) acc[t][i] *= corr[i];

        __builtin_amdgcn_wave_barrier();   // keep ds_store(P) before ds_load(P)

        // O += P(16x32) x V(32x64): A from per-wave LDS, B gathered from Vs
        v16bf pa;
        #pragma unroll
        for (int j = 0; j < 8; ++j) {
            pa[j]     = myP[lr * 34 + koff + j];
            pa[j + 8] = myP[lr * 34 + 16 + koff + j];
        }
        #pragma unroll
        for (int t = 0; t < 4; ++t) {
            v16bf vb;
            const int col = t * 16 + lr;
            #pragma unroll
            for (int j = 0; j < 8; ++j) {
                vb[j]     = Vs[(koff + j) * 66 + col];
                vb[j + 8] = Vs[(16 + koff + j) * 66 + col];
            }
            acc[t] = __builtin_amdgcn_wmma_f32_16x16x32_bf16(
                false, pa, false, vb, (short)0, acc[t], false, false);
        }
    }

    // out[((b*S + s)*H + h)*64 + d] = acc / l
    #pragma unroll
    for (int t = 0; t < 4; ++t) {
        #pragma unroll
        for (int i = 0; i < 8; ++i) {
            const int row16 = (lane < 16) ? i : (i + 8);
            const int s = q0 + row16;
            const int d = t * 16 + lr;
            out[(((size_t)(b * Sn + s)) * Hn + h) * Dn + d] = acc[t][i] / l_run[i];
        }
    }
}

extern "C" void kernel_launch(void* const* d_in, const int* in_sizes, int n_in,
                              void* d_out, int out_size, void* d_ws, size_t ws_size,
                              hipStream_t stream) {
    const float* x  = (const float*)d_in[0];
    const float* Wq = (const float*)d_in[1];
    const float* bq = (const float*)d_in[2];
    const float* Wk = (const float*)d_in[3];
    const float* bk = (const float*)d_in[4];
    const float* Wv = (const float*)d_in[5];
    const float* bv = (const float*)d_in[6];
    float* out = (float*)d_out;

    const size_t per = (size_t)Bn * Hn * Sn * Dn;   // 8,388,608 elems (16MB bf16)
    __bf16* Qw = (__bf16*)d_ws;
    __bf16* Kw = Qw + per;
    __bf16* Vw = Kw + per;

    dim3 gProj(64, 16, 3);     // (8192/128, 1024/64, {Q,K,V})
    qkv_proj_kernel<<<gProj, 256, 0, stream>>>(x, Wq, bq, Wk, bk, Wv, bv, Qw, Kw, Vw);

    dim3 gAttn(Sn / 128, Bn * Hn);   // (16, 64)
    flash_attn_kernel<<<gAttn, 256, 0, stream>>>(Qw, Kw, Vw, out);
}
